// Convolution_12171937317096
// MI455X (gfx1250) — matrix operand order repaired and checked by reading, not compiled, not run on hardware
//
#include <hip/hip_runtime.h>
#include <hip/hip_bf16.h>

typedef __attribute__((ext_vector_type(16))) __bf16 v16bf;
typedef __attribute__((ext_vector_type(8)))  __bf16 v8bf;
typedef __attribute__((ext_vector_type(4)))  __bf16 v4bf;
typedef __attribute__((ext_vector_type(8)))  float  v8f;
typedef __attribute__((ext_vector_type(4)))  int    v4i;
typedef __attribute__((address_space(1))) v4i as1_v4i;
typedef __attribute__((address_space(3))) v4i as3_v4i;

#define NRB 8

// ---------------------------------------------------------------------------
// Pass 0: one-shot f32 -> bf16 conversion of x (removes 9x-redundant VALU
// conversion work from the conv kernel and halves global read bytes).
// ---------------------------------------------------------------------------
__global__ void __launch_bounds__(256)
convert_bf16(const float* __restrict__ x, __bf16* __restrict__ xb) {
  int gid = blockIdx.x * blockDim.x + threadIdx.x;  // one float4 per thread
  const float4* p = (const float4*)x + gid;
  float4 f = *p;
  v4bf v;
  v[0] = (__bf16)f.x;
  v[1] = (__bf16)f.y;
  v[2] = (__bf16)f.z;
  v[3] = (__bf16)f.w;
  *((v4bf*)xb + gid) = v;
}

// ---------------------------------------------------------------------------
// Pass 1: build effective conv kernel K'[27][64][64] (self-connection folded
// into center tap t=13), bf16, pre-swizzled into WMMA B-fragment layout:
// frag f = ((t*2+kb)*4+nb); lane = (n&15) + 16*(((k&31)>>3)&1); element = (k&7) + 8*((k&31)>>4).
// ---------------------------------------------------------------------------
__global__ void build_kernel(const float* __restrict__ Wsc0,
                             const float* __restrict__ Wsc1,
                             const float* __restrict__ w1,
                             const float* __restrict__ w2,
                             const float* __restrict__ w3,
                             const float* __restrict__ w4,
                             __bf16* __restrict__ Kswz) {
  int gid = blockIdx.x * blockDim.x + threadIdx.x;
  if (gid >= 27 * 64 * 64) return;
  int n = gid & 63;
  int k = (gid >> 6) & 63;
  int t = gid >> 12;
  int iz = t % 3, iy = (t / 3) % 3, ix = t / 9;
  float lx = (float)(ix - 1), ly = (float)(iy - 1), lz = (float)(iz - 1);
  float d = sqrtf(lx * lx + ly * ly + lz * lz);

  float emb[NRB];
  const float step = 1.5f / 9.0f;
  #pragma unroll
  for (int r = 0; r < NRB; ++r) {
    float diff = (d - (float)(r + 1) * step) / step;
    float q = diff * diff;
    emb[r] = (q < 1.0f) ? 1.14136f * expf(2.0f) * expf(-2.0f / (1.0f - q))
                        : 0.0f;
  }
  float invd = 1.0f / fmaxf(d, 1e-12f);
  float sh1v[3] = {1.7320508075688772f * lx * invd,
                   1.7320508075688772f * ly * invd,
                   1.7320508075688772f * lz * invd};

  const float alpha = 0.17677669529663687f;  // 1/sqrt(2*MUL)
  const float inv27 = 1.0f / 27.0f;
  auto Wsum = [&](const float* w, int a, int b) {
    float s = 0.0f;
    #pragma unroll
    for (int r = 0; r < NRB; ++r) s += emb[r] * w[(r * 16 + a) * 16 + b];
    return s * inv27;
  };

  bool center = (t == 13);
  float v = 0.0f;
  if (k < 16 && n < 16) {
    v = alpha * Wsum(w1, k, n);
    if (center) v += 0.25f * Wsc0[k * 16 + n];
  } else if (k < 16) {
    int o = (n - 16) / 3, m = (n - 16) % 3;
    v = alpha * Wsum(w2, k, o) * sh1v[m];
  } else if (n < 16) {
    int i = (k - 16) / 3, m = (k - 16) % 3;
    v = (alpha * 0.5773502691896258f) * Wsum(w4, i, n) * sh1v[m];
  } else {
    int i = (k - 16) / 3, mi = (k - 16) % 3;
    int o = (n - 16) / 3, mo = (n - 16) % 3;
    if (mi == mo) {
      v = alpha * Wsum(w3, i, o);
      if (center) v += 0.25f * Wsc1[i * 16 + o];
    }
  }

  int kb = k >> 5, kk = k & 31;
  int nb = n >> 4, nn = n & 15;
  int lane = nn + 16 * ((kk >> 3) & 1);
  int e = (kk & 7) + 8 * (kk >> 4);
  Kswz[(((t * 2 + kb) * 4 + nb) * 32 + lane) * 16 + e] = (__bf16)v;
}

// ---------------------------------------------------------------------------
// Pass 2: 27-tap conv as shifted 16x64 @ 64x64 GEMMs on v_wmma_f32_16x16x32_bf16.
// One 128-thread block per (b,x,y) row; LDS = 9 neighbor rows x 34 z-slots of
// bf16 channels. ASYNC path fills LDS with global_load_async_to_lds_b128.
// ---------------------------------------------------------------------------
template <bool ASYNC>
__global__ void __launch_bounds__(128)
conv_wmma(const float* __restrict__ x, const __bf16* __restrict__ xb,
          const __bf16* __restrict__ Kswz, float* __restrict__ out) {
  __shared__ alignas(16) __bf16 tile[9 * 34 * 64];  // 38.25 KB

  const int tid = threadIdx.x;
  const int bidx = blockIdx.x;
  const int yy = bidx & 31;
  const int xx = (bidx >> 5) & 31;
  const int bb = bidx >> 10;

  if (ASYNC) {
    // ---- async DMA fill: 16-byte chunks, halos zeroed via ds stores ------
    for (int i = tid; i < 9 * 34 * 8; i += 128) {
      int ch = i & 7;        // 16-byte chunk within the 128-byte channel row
      int rz = i >> 3;
      int zslot = rz % 34;
      int row = rz / 34;
      int gx = xx + row / 3 - 1;
      int gy = yy + row % 3 - 1;
      int gz = zslot - 1;
      __bf16* lp = tile + (row * 34 + zslot) * 64 + ch * 8;
      bool valid =
          ((unsigned)gx < 32u) && ((unsigned)gy < 32u) && ((unsigned)gz < 32u);
      if (valid) {
        const __bf16* gp =
            xb + (((((bb * 32 + gx) * 32 + gy) * 32 + gz) * 64) + ch * 8);
#if __has_builtin(__builtin_amdgcn_global_load_async_to_lds_b128)
        // signature per hipcc diagnostic: (v4i AS1*, v4i AS3*, imm, imm)
        __builtin_amdgcn_global_load_async_to_lds_b128(
            (as1_v4i*)(size_t)gp, (as3_v4i*)(size_t)lp, 0, 0);
#else
        unsigned loff = (unsigned)(size_t)lp;           // LDS byte offset
        unsigned long long ga = (unsigned long long)(size_t)gp;
        asm volatile("global_load_async_to_lds_b128 %0, %1, off"
                     :: "v"(loff), "v"(ga)
                     : "memory");
#endif
      } else {
        *(v8bf*)lp = (v8bf){};  // zero halo (disjoint addresses, DS pipe)
      }
    }
#if __has_builtin(__builtin_amdgcn_s_wait_asynccnt)
    __builtin_amdgcn_s_wait_asynccnt(0);
#else
    asm volatile("s_wait_asynccnt 0x0" ::: "memory");
#endif
    __syncthreads();
  } else {
    // ---- fallback fill: convert f32 -> bf16 in-flight --------------------
    for (int i = tid; i < 9 * 34 * 16; i += 128) {
      int c4 = i & 15;
      int rz = i >> 4;
      int zslot = rz % 34;
      int row = rz / 34;
      int gx = xx + row / 3 - 1;
      int gy = yy + row % 3 - 1;
      int gz = zslot - 1;
      v4bf val = {};
      if ((unsigned)gx < 32u && (unsigned)gy < 32u && (unsigned)gz < 32u) {
        const float4* p =
            (const float4*)(x + ((((bb * 32 + gx) * 32 + gy) * 32 + gz) * 64)) +
            c4;
        float4 f = *p;
        val[0] = (__bf16)f.x;
        val[1] = (__bf16)f.y;
        val[2] = (__bf16)f.z;
        val[3] = (__bf16)f.w;
      }
      *(v4bf*)(tile + (row * 34 + zslot) * 64 + c4 * 4) = val;
    }
    __syncthreads();
  }

  // ---- WMMA compute ------------------------------------------------------
  const int lane = tid & 31;
  const int wv = tid >> 5;
  const int mt = wv >> 1;   // z half: voxels mt*16 .. mt*16+15
  const int nh = wv & 1;    // output-channel half: 32*nh .. 32*nh+31
  const int g = lane >> 4;
  const int m = lane & 15;

  v8f acc0 = {};
  v8f acc1 = {};
  const v16bf* __restrict__ Kv = (const v16bf*)Kswz;
  const int zv = mt * 16 + m;
  const int c0 = 8 * g;

  #pragma unroll 9
  for (int t = 0; t < 27; ++t) {
    int dz = t % 3 - 1;
    int row = t / 3;

    // A fragments (16-bit A layout): lane half g holds k in
    // [8g,8g+7] U [8g+16,8g+23] => two ds_load_b128 chunks per fragment.
    const __bf16* abase = tile + (row * 34 + (zv + dz + 1)) * 64;
    v8bf a0lo = *(const v8bf*)(abase + c0);
    v8bf a0hi = *(const v8bf*)(abase + c0 + 16);
    v8bf a1lo = *(const v8bf*)(abase + 32 + c0);
    v8bf a1hi = *(const v8bf*)(abase + 32 + c0 + 16);
    v16bf a0 = __builtin_shufflevector(a0lo, a0hi, 0, 1, 2, 3, 4, 5, 6, 7, 8,
                                       9, 10, 11, 12, 13, 14, 15);
    v16bf a1 = __builtin_shufflevector(a1lo, a1hi, 0, 1, 2, 3, 4, 5, 6, 7, 8,
                                       9, 10, 11, 12, 13, 14, 15);

    // B fragments: contiguous 32B per lane, L2-resident (221 KB total)
    int fb = (t * 2) * 4 + 2 * nh;
    v16bf b00 = Kv[(fb + 0) * 32 + lane];
    v16bf b01 = Kv[(fb + 1) * 32 + lane];
    v16bf b10 = Kv[(fb + 4) * 32 + lane];
    v16bf b11 = Kv[(fb + 5) * 32 + lane];

    acc0 = __builtin_amdgcn_wmma_f32_16x16x32_bf16(false, a0, false, b00,
                                                   (short)0, acc0, false, false);
    acc0 = __builtin_amdgcn_wmma_f32_16x16x32_bf16(false, a1, false, b10,
                                                   (short)0, acc0, false, false);
    acc1 = __builtin_amdgcn_wmma_f32_16x16x32_bf16(false, a0, false, b01,
                                                   (short)0, acc1, false, false);
    acc1 = __builtin_amdgcn_wmma_f32_16x16x32_bf16(false, a1, false, b11,
                                                   (short)0, acc1, false, false);
  }

  // ---- store: C/D layout M = j + 8*g, N = lane&15 ------------------------
  float* __restrict__ obase =
      out + (size_t)(((bb * 32 + xx) * 32 + yy) * 32) * 64;
  #pragma unroll
  for (int j = 0; j < 8; ++j) {
    int z = mt * 16 + j + 8 * g;
    int n0 = 32 * nh + m;
    obase[z * 64 + n0] = acc0[j];
    obase[z * 64 + n0 + 16] = acc1[j];
  }
}

// ---------------------------------------------------------------------------
extern "C" void kernel_launch(void* const* d_in, const int* in_sizes, int n_in,
                              void* d_out, int out_size, void* d_ws,
                              size_t ws_size, hipStream_t stream) {
  const float* x    = (const float*)d_in[0];
  const float* Wsc0 = (const float*)d_in[1];
  const float* Wsc1 = (const float*)d_in[2];
  const float* w1   = (const float*)d_in[3];
  const float* w2   = (const float*)d_in[4];
  const float* w3   = (const float*)d_in[5];
  const float* w4   = (const float*)d_in[6];

  const size_t needK  = (size_t)27 * 64 * 64 * sizeof(__bf16);   // 221,184 B
  const size_t nElem  = (size_t)4 * 32 * 32 * 32 * 64;           // 8.4M
  const size_t needXb = nElem * sizeof(__bf16);                  // 16.8 MB

  __bf16* Kswz = (__bf16*)d_ws;
  __bf16* xb   = (__bf16*)((char*)d_ws + needK);
  bool use_async = (ws_size >= needK + needXb);

  build_kernel<<<(27 * 64 * 64 + 255) / 256, 256, 0, stream>>>(
      Wsc0, Wsc1, w1, w2, w3, w4, Kswz);

  if (use_async) {
    convert_bf16<<<(int)(nElem / 4 / 256), 256, 0, stream>>>(x, xb);
    conv_wmma<true><<<4 * 32 * 32, 128, 0, stream>>>(x, xb, Kswz, (float*)d_out);
  } else {
    conv_wmma<false><<<4 * 32 * 32, 128, 0, stream>>>(x, xb, Kswz, (float*)d_out);
  }
}
